// DrugEncoder_45457933861191
// MI455X (gfx1250) — matrix operand order repaired and checked by reading, not compiled, not run on hardware
//
#include <hip/hip_runtime.h>
#include <hip/hip_bf16.h>
#include <math.h>

typedef _Float16 h16;
typedef __attribute__((ext_vector_type(16))) _Float16 v16h;
typedef __attribute__((ext_vector_type(8)))  _Float16 v8h;
typedef __attribute__((ext_vector_type(8)))  float    v8f;
typedef __attribute__((ext_vector_type(4)))  float    v4f;

#define NN 100000
#define DD 128
#define LL 3
#define GG 2048
#define EE 1600000
#define BN_EPS 1e-5f

// ---------------------------------------------------------------------------
// agg[i][:] = h[i][:]   (h row stride given in float4 units)
// ---------------------------------------------------------------------------
__global__ void k_init_agg(const float* __restrict__ h, int hStride4,
                           float* __restrict__ agg) {
  long long gid = (long long)blockIdx.x * 256 + threadIdx.x;
  if (gid < (long long)NN * 32) {
    long long i = gid >> 5;
    int c4 = (int)(gid & 31);
    ((v4f*)agg)[i * 32 + c4] = ((const v4f*)h)[i * hStride4 + c4];
  }
}

// ---------------------------------------------------------------------------
// agg[dst[e]][:] += h[src[e]][:]  — edge-parallel, 4 floats per thread
// ---------------------------------------------------------------------------
__global__ void k_scatter_add(const int* __restrict__ src, const int* __restrict__ dst,
                              const float* __restrict__ h, int hStride4,
                              float* __restrict__ agg) {
  long long gid = (long long)blockIdx.x * 256 + threadIdx.x;
  if (gid < (long long)EE * 32) {
    long long e = gid >> 5;
    int c4 = (int)(gid & 31);
    int s = src[e];
    int d = dst[e];
    v4f v = ((const v4f*)h)[(long long)s * hStride4 + c4];
    float* ap = agg + (long long)d * DD + c4 * 4;
    atomicAdd(ap + 0, v.x);
    atomicAdd(ap + 1, v.y);
    atomicAdd(ap + 2, v.z);
    atomicAdd(ap + 3, v.w);
  }
}

// ---------------------------------------------------------------------------
// Convert this layer's W1/W2 (f32 row-major [k][n]) to f16 column-major
// wt[n*128+k] so the WMMA B-fragment is a contiguous 32B LDS read per lane.
// ---------------------------------------------------------------------------
__global__ void k_prep_weights(const float* __restrict__ W1, const float* __restrict__ W2,
                               h16* __restrict__ w1t, h16* __restrict__ w2t) {
  int e = blockIdx.x * 256 + threadIdx.x;
  if (e < DD * DD) {
    int k = e & 127;
    int n = e >> 7;
    w1t[e] = (h16)W1[k * DD + n];
    w2t[e] = (h16)W2[k * DD + n];
  }
}

__global__ void k_zero_stats(float* __restrict__ s) {
  int t = blockIdx.x * 256 + threadIdx.x;
  if (t < 256) s[t] = 0.f;   // sum[128] + sumsq[128]
}

// ---------------------------------------------------------------------------
// Fused MLP:  m = relu( relu(agg @ W1 + b1) @ W2 + b2 )
// Writes m into rep slot (row stride 384) and accumulates BN sum / sumsq.
// Block = 256 threads = 8 waves; wave owns 16 rows x 128 cols.
// B fragments of a whole k-step are preloaded as a register array so the
// 16 ds_load_b128 issue as a clause ahead of the 8 back-to-back v_wmma.
// ---------------------------------------------------------------------------
__launch_bounds__(256)
__global__ void k_mlp_wmma(const float* __restrict__ agg,
                           const h16* __restrict__ w1t, const h16* __restrict__ w2t,
                           const float* __restrict__ b1, const float* __restrict__ b2,
                           float* __restrict__ repSlot,
                           float* __restrict__ gsum, float* __restrict__ gsq) {
  __shared__ h16 Ws[DD * DD];       // 32 KB: W1 then (after barrier) W2, col-major f16
  __shared__ h16 inter[8][16 * DD]; // 32 KB: per-wave 16x128 f16 intermediate

  const int tid  = threadIdx.x;
  const int wave = tid >> 5;
  const int lane = tid & 31;
  const int ln   = lane & 15;   // row (A/C) or col (B/C) within tile
  const int hi   = lane >> 4;   // K-half selector / row-half selector
  const int rowBase = blockIdx.x * 128 + wave * 16;

  int r0 = rowBase + ln;
  if (r0 >= NN) r0 = NN - 1;    // clamp (stores masked later)
  const float* aRow = agg + (long long)r0 * DD;

  // prefetch this wave's A row while weights are being staged
  __builtin_prefetch(aRow, 0, 1);
  __builtin_prefetch(aRow + 64, 0, 1);

  // ---- stage W1 into LDS (contiguous copy, already col-major f16) ----
  {
    const v8h* s = (const v8h*)w1t;
    v8h* d = (v8h*)Ws;
    for (int i = tid; i < DD * DD / 8; i += 256) d[i] = s[i];
  }
  __syncthreads();

  // ---- GEMM1: acc = agg @ W1 ----
  v8f acc[8] = {};
#pragma unroll
  for (int k0 = 0; k0 < DD; k0 += 32) {
    // A-frag: lane holds row r0, K chunk [k0+hi*16, +16)
    const float* p0 = aRow + k0 + hi * 16;
    v4f f0 = *(const v4f*)(p0);
    v4f f1 = *(const v4f*)(p0 + 4);
    v4f f2 = *(const v4f*)(p0 + 8);
    v4f f3 = *(const v4f*)(p0 + 12);
    v16h a;
#pragma unroll
    for (int j = 0; j < 4; ++j) {
      a[j]      = (h16)f0[j];
      a[4 + j]  = (h16)f1[j];
      a[8 + j]  = (h16)f2[j];
      a[12 + j] = (h16)f3[j];
    }
    // preload all B-frags of this k-step, then run the 8 WMMAs back-to-back
    v16h bf[8];
#pragma unroll
    for (int t = 0; t < 8; ++t)
      bf[t] = *(const v16h*)&Ws[(t * 16 + ln) * DD + k0 + hi * 16];
#pragma unroll
    for (int t = 0; t < 8; ++t)
      acc[t] = __builtin_amdgcn_wmma_f32_16x16x32_f16(
          false, a, false, bf[t], (short)0, acc[t], false, false);
  }

  // ---- bias1 + relu -> per-wave f16 intermediate (row-major 16x128) ----
  h16* myInter = inter[wave];
#pragma unroll
  for (int t = 0; t < 8; ++t) {
    float bias = b1[t * 16 + ln];
#pragma unroll
    for (int v = 0; v < 8; ++v) {
      float x = acc[t][v] + bias;
      x = x > 0.f ? x : 0.f;
      myInter[(hi * 8 + v) * DD + t * 16 + ln] = (h16)x;
    }
  }
  __syncthreads();                 // all waves done reading W1

  // ---- stage W2 over W1 ----
  {
    const v8h* s = (const v8h*)w2t;
    v8h* d = (v8h*)Ws;
    for (int i = tid; i < DD * DD / 8; i += 256) d[i] = s[i];
  }
  __syncthreads();

  // ---- GEMM2: acc2 = inter @ W2 ----
  v8f acc2[8] = {};
#pragma unroll
  for (int k0 = 0; k0 < DD; k0 += 32) {
    v8h lo = *(const v8h*)&myInter[ln * DD + k0 + hi * 16];
    v8h hh = *(const v8h*)&myInter[ln * DD + k0 + hi * 16 + 8];
    v16h a = __builtin_shufflevector(lo, hh, 0, 1, 2, 3, 4, 5, 6, 7,
                                     8, 9, 10, 11, 12, 13, 14, 15);
    v16h bf[8];
#pragma unroll
    for (int t = 0; t < 8; ++t)
      bf[t] = *(const v16h*)&Ws[(t * 16 + ln) * DD + k0 + hi * 16];
#pragma unroll
    for (int t = 0; t < 8; ++t)
      acc2[t] = __builtin_amdgcn_wmma_f32_16x16x32_f16(
          false, a, false, bf[t], (short)0, acc2[t], false, false);
  }

  // ---- bias2 + relu, store m, accumulate BN partials ----
  float sumP[8], sqP[8];
#pragma unroll
  for (int t = 0; t < 8; ++t) {
    float bias = b2[t * 16 + ln];
    float sp = 0.f, qp = 0.f;
#pragma unroll
    for (int v = 0; v < 8; ++v) {
      int r = rowBase + hi * 8 + v;
      float x = acc2[t][v] + bias;
      x = x > 0.f ? x : 0.f;
      if (r < NN) {
        repSlot[(long long)r * (LL * DD) + t * 16 + ln] = x;
        sp += x;
        qp += x * x;
      }
    }
    sumP[t] = sp;
    sqP[t] = qp;
  }

  // ---- block-level BN reduction in LDS (reuse inter area), then global ----
  __syncthreads();                 // inter no longer needed
  float* statL = (float*)&inter[0][0];   // 256 floats: sum[128], sq[128]
  statL[tid] = 0.f;
  __syncthreads();
#pragma unroll
  for (int t = 0; t < 8; ++t) {
    atomicAdd(&statL[t * 16 + ln], sumP[t]);
    atomicAdd(&statL[128 + t * 16 + ln], sqP[t]);
  }
  __syncthreads();
  if (tid < 128) {
    atomicAdd(&gsum[tid], statL[tid]);
    atomicAdd(&gsq[tid], statL[128 + tid]);
  }
}

// ---------------------------------------------------------------------------
// BN finalize + apply (training-mode batch stats, biased variance)
// ---------------------------------------------------------------------------
__global__ void k_bn_finalize(const float* __restrict__ gsum, const float* __restrict__ gsq,
                              float* __restrict__ mean, float* __restrict__ inv) {
  int c = threadIdx.x;
  if (c < DD) {
    float m = gsum[c] * (1.0f / NN);
    float var = gsq[c] * (1.0f / NN) - m * m;
    mean[c] = m;
    inv[c] = rsqrtf(var + BN_EPS);
  }
}

__global__ void k_bn_apply(float* __restrict__ repSlot,
                           const float* __restrict__ mean, const float* __restrict__ inv,
                           const float* __restrict__ gamma, const float* __restrict__ beta) {
  long long gid = (long long)blockIdx.x * 256 + threadIdx.x;
  if (gid < (long long)NN * 32) {
    long long i = gid >> 5;
    int c4 = (int)(gid & 31);
    int c = c4 * 4;
    v4f v = *(v4f*)&repSlot[i * (LL * DD) + c];
    v4f mn = *(const v4f*)&mean[c];
    v4f iv = *(const v4f*)&inv[c];
    v4f gm = *(const v4f*)&gamma[c];
    v4f bt = *(const v4f*)&beta[c];
    v4f r;
    r.x = gm.x * (v.x - mn.x) * iv.x + bt.x;
    r.y = gm.y * (v.y - mn.y) * iv.y + bt.y;
    r.z = gm.z * (v.z - mn.z) * iv.z + bt.z;
    r.w = gm.w * (v.w - mn.w) * iv.w + bt.w;
    *(v4f*)&repSlot[i * (LL * DD) + c] = r;
  }
}

// ---------------------------------------------------------------------------
// Segment max via monotone uint encoding + atomicMax
// ---------------------------------------------------------------------------
__device__ __forceinline__ unsigned enc_f32(float x) {
  unsigned u = __float_as_uint(x);
  return (u & 0x80000000u) ? ~u : (u | 0x80000000u);
}

__global__ void k_seg_init(unsigned* __restrict__ outU) {
  int gid = blockIdx.x * 256 + threadIdx.x;
  if (gid < GG * LL * DD) outU[gid] = 0x007FFFFFu;   // enc(-inf)
}

__global__ void k_seg_max(const float* __restrict__ rep, const int* __restrict__ batch,
                          unsigned* __restrict__ outU) {
  int i = blockIdx.x;            // node
  int c = threadIdx.x;           // 0..383
  int g = batch[i];
  unsigned e = enc_f32(rep[(long long)i * (LL * DD) + c]);
  atomicMax(&outU[(long long)g * (LL * DD) + c], e);
}

__global__ void k_seg_decode(unsigned* __restrict__ outU) {
  int gid = blockIdx.x * 256 + threadIdx.x;
  if (gid < GG * LL * DD) {
    unsigned u = outU[gid];
    float f = (u & 0x80000000u) ? __uint_as_float(u & 0x7FFFFFFFu)
                                : __uint_as_float(~u);
    ((float*)outU)[gid] = f;
  }
}

// ---------------------------------------------------------------------------
extern "C" void kernel_launch(void* const* d_in, const int* in_sizes, int n_in,
                              void* d_out, int out_size, void* d_ws, size_t ws_size,
                              hipStream_t stream) {
  const float* x     = (const float*)d_in[0];
  const int*   edge  = (const int*)d_in[1];   // [2, E] (doc: integer -> const int*)
  const int*   batch = (const int*)d_in[2];   // [N]
  const float* W1    = (const float*)d_in[3]; // [L,128,128]
  const float* b1    = (const float*)d_in[4]; // [L,128]
  const float* W2    = (const float*)d_in[5];
  const float* b2    = (const float*)d_in[6];
  const float* gamma = (const float*)d_in[7];
  const float* beta  = (const float*)d_in[8];

  const int* srcIdx = edge;
  const int* dstIdx = edge + EE;

  // workspace layout
  char* ws = (char*)d_ws;
  const size_t REP_BYTES = (size_t)NN * LL * DD * sizeof(float);   // 153.6 MB
  const size_t AGG_BYTES = (size_t)NN * DD * sizeof(float);        //  51.2 MB
  float* rep = (float*)ws;
  float* agg = (float*)(ws + REP_BYTES);
  h16*   w1t = (h16*)(ws + REP_BYTES + AGG_BYTES);
  h16*   w2t = w1t + DD * DD;
  float* stats = (float*)(w2t + DD * DD);   // sum[128], sq[128], mean[128], inv[128]
  float* gsum = stats;
  float* gsq  = stats + 128;
  float* meanB = stats + 256;
  float* invB  = stats + 384;

  const int nCopy    = (NN * 32 + 255) / 256;
  const int nScatter = (int)(((long long)EE * 32 + 255) / 256);
  const int nMlp     = (NN + 127) / 128;
  const int nBn      = (NN * 32 + 255) / 256;
  const int nOut     = (GG * LL * DD + 255) / 256;

  for (int l = 0; l < LL; ++l) {
    const float* h = (l == 0) ? x : (rep + (l - 1) * DD);
    int hStride4 = (l == 0) ? (DD / 4) : (LL * DD / 4);

    k_init_agg<<<nCopy, 256, 0, stream>>>(h, hStride4, agg);
    k_scatter_add<<<nScatter, 256, 0, stream>>>(srcIdx, dstIdx, h, hStride4, agg);
    k_prep_weights<<<(DD * DD + 255) / 256, 256, 0, stream>>>(
        W1 + (size_t)l * DD * DD, W2 + (size_t)l * DD * DD, w1t, w2t);
    k_zero_stats<<<1, 256, 0, stream>>>(gsum);
    k_mlp_wmma<<<nMlp, 256, 0, stream>>>(agg, w1t, w2t,
                                         b1 + l * DD, b2 + l * DD,
                                         rep + l * DD, gsum, gsq);
    k_bn_finalize<<<1, 128, 0, stream>>>(gsum, gsq, meanB, invB);
    k_bn_apply<<<nBn, 256, 0, stream>>>(rep + l * DD, meanB, invB,
                                        gamma + l * DD, beta + l * DD);
  }

  unsigned* outU = (unsigned*)d_out;
  k_seg_init<<<nOut, 256, 0, stream>>>(outU);
  k_seg_max<<<NN, LL * DD, 0, stream>>>(rep, batch, outU);
  k_seg_decode<<<nOut, 256, 0, stream>>>(outU);

  (void)in_sizes; (void)n_in; (void)out_size; (void)ws_size;
}
